// TemporalContrastiveLoss_62182536512268
// MI455X (gfx1250) — matrix operand order repaired and checked by reading, not compiled, not run on hardware
//
#include <hip/hip_runtime.h>
#include <hip/hip_bf16.h>

// ---- problem constants (match reference) ----
#define NN      8192
#define D_EMB   128
#define PP      256
#define KK      256
#define DEG     32
#define FF      16
#define TEMP_INV (1.0f / 0.07f)

typedef _Float16 v16h __attribute__((ext_vector_type(16)));
typedef _Float16 v8h  __attribute__((ext_vector_type(8)));
typedef float    v8f  __attribute__((ext_vector_type(8)));

// ---------- helpers ----------
__device__ __forceinline__ float waveSum(float v) {
#pragma unroll
    for (int o = 16; o > 0; o >>= 1) v += __shfl_xor(v, o, 32);
    return v;
}

__device__ __forceinline__ float time_enc(float dt, const float* __restrict__ om,
                                          const float* __restrict__ ph) {
    float r = om[0] * dt + ph[0];
#pragma unroll
    for (int f = 1; f < FF; ++f) r += __sinf(om[f] * dt + ph[f]);
    return r;
}

// ---------- K1: row L2-normalize z -> f16 copy in workspace ----------
// one wave per row; 8 rows / 256-thread block
__global__ __launch_bounds__(256) void normalize_kernel(const float* __restrict__ z,
                                                        _Float16* __restrict__ zh) {
    int row  = blockIdx.x * 8 + (threadIdx.x >> 5);
    int lane = threadIdx.x & 31;
    const float4 v = *(const float4*)(z + (size_t)row * D_EMB + lane * 4);
    float ss = v.x * v.x + v.y * v.y + v.z * v.z + v.w * v.w;
    ss = waveSum(ss);
    float inv = 1.0f / fmaxf(sqrtf(ss), 1e-12f);
    _Float16* p = zh + (size_t)row * D_EMB + lane * 4;
    p[0] = (_Float16)(v.x * inv);
    p[1] = (_Float16)(v.y * inv);
    p[2] = (_Float16)(v.z * inv);
    p[3] = (_Float16)(v.w * inv);
}

// ---------- K2: term1 via WMMA ----------
// mu_iy[256x32] = 2 * (Z_nb[256x128] x Z_Nx^T[128x32]) - 2   (unit rows)
// Single workgroup, 8 waves, 4 C-tiles (16x16) per wave, K=128 in 4 WMMA steps.
__global__ __launch_bounds__(256) void term1_kernel(
    const _Float16* __restrict__ zh,
    const float* __restrict__ edge_times,
    const float* __restrict__ current_time,
    const float* __restrict__ omega,
    const float* __restrict__ phi,
    const int* __restrict__ query_idx,
    const int* __restrict__ neighbor_idxs,
    const int* __restrict__ neighbors,
    float* __restrict__ term1_out) {
    __shared__ float muS[PP][DEG];   // 32 KB
    __shared__ float teS[DEG];
    __shared__ int   nbS[PP];
    __shared__ int   nqS[DEG];

    const int tid = threadIdx.x;
    const int q   = query_idx[0];
    nbS[tid] = neighbor_idxs[tid];
    if (tid < DEG) nqS[tid] = neighbors[q * DEG + tid];
    __syncthreads();

    const int wave  = tid >> 5;
    const int lane  = tid & 31;
    const int lhalf = lane >> 4;   // 0|1
    const int l15   = lane & 15;

#pragma unroll
    for (int t = 0; t < 4; ++t) {
        const int tile  = wave + t * 8;      // 0..31 unique (mtile,ntile) pairs
        const int mtile = tile >> 1;
        const int ntile = tile & 1;

        const int aRow = nbS[mtile * 16 + l15];
        const int bRow = nqS[ntile * 16 + l15];
        // A: lanes 0-15 hold K in [ks*32, ks*32+8)∪[ks*32+16, ks*32+24); lanes 16-31 shifted +8
        const _Float16* aBase = zh + (size_t)aRow * D_EMB + lhalf * 8;
        // B: lanes 0-15 hold K in [ks*32, ks*32+8)∪[ks*32+16, ks*32+24); lanes 16-31 shifted +8
        const _Float16* bBase = zh + (size_t)bRow * D_EMB + lhalf * 8;
        __builtin_prefetch(bBase, 0, 0);     // -> global_prefetch_b8

        v8f c = {};
#pragma unroll
        for (int ks = 0; ks < 4; ++ks) {
            const v8h a0 = *(const v8h*)(aBase + ks * 32);
            const v8h a1 = *(const v8h*)(aBase + ks * 32 + 16);
            const v8h b0 = *(const v8h*)(bBase + ks * 32);
            const v8h b1 = *(const v8h*)(bBase + ks * 32 + 16);
            v16h a, b;
#pragma unroll
            for (int e = 0; e < 8; ++e) {
                a[e] = a0[e];  a[e + 8] = a1[e];
                b[e] = b0[e];  b[e + 8] = b1[e];
            }
            c = __builtin_amdgcn_wmma_f32_16x16x32_f16(
                    false, a, false, b, (short)0, c, false, false);
        }
        // C layout: VGPR r -> M = mtile*16 + r + 8*lhalf, N = ntile*16 + (lane&15)
#pragma unroll
        for (int r = 0; r < 8; ++r) {
            muS[mtile * 16 + r + lhalf * 8][ntile * 16 + l15] = 2.0f * c[r] - 2.0f;
        }
    }
    __syncthreads();

    if (tid < DEG) {
        float dt = current_time[0] - edge_times[(size_t)q * NN + nqS[tid]];
        teS[tid] = time_enc(dt, omega, phi);
    }
    __syncthreads();

    // thread tid owns row tid: softmax-weighted sum over DEG
    float wv[DEG];
    float s = 0.0f;
#pragma unroll
    for (int j = 0; j < DEG; ++j) {
        float m = muS[tid][j];
        float w = __expf(m * TEMP_INV - teS[j]);
        wv[j] = w;
        s += w;
    }
    float inv = 1.0f / (s + 1e-8f);
    float t1 = 0.0f;
#pragma unroll
    for (int j = 0; j < DEG; ++j) t1 += wv[j] * inv * muS[tid][j];
    term1_out[tid] = t1;
}

// ---------- K3: term2 + per-row loss partials; one wave per row i ----------
__global__ __launch_bounds__(32) void term2_kernel(
    const _Float16* __restrict__ zh,
    const float* __restrict__ edge_times,
    const float* __restrict__ current_time,
    const float* __restrict__ core_values,
    const float* __restrict__ omega,
    const float* __restrict__ phi,
    const int* __restrict__ query_idx,
    const int* __restrict__ neighbor_idxs,
    const int* __restrict__ neighbors,
    const float* __restrict__ term1,
    float* __restrict__ rowpart) {
    const int i    = blockIdx.x;
    const int lane = threadIdx.x;
    const int q    = query_idx[0];
    const int nb   = neighbor_idxs[i];

    // query row, 4 floats per lane
    float zq[4];
#pragma unroll
    for (int k = 0; k < 4; ++k) zq[k] = (float)zh[(size_t)q * D_EMB + lane * 4 + k];

    // lane "lane" owns j = lane: its neighbor row + time encoding
    const int  r  = neighbors[nb * DEG + lane];
    const float dt = current_time[0] - edge_times[(size_t)nb * NN + r];
    const float te = time_enc(dt, omega, phi);

    // cooperative exact mu_ix for each j
    float muj = 0.0f;
    for (int j = 0; j < DEG; ++j) {
        int rj = __shfl(r, j, 32);
        float ss = 0.0f;
#pragma unroll
        for (int k = 0; k < 4; ++k) {
            float d = (float)zh[(size_t)rj * D_EMB + lane * 4 + k] - zq[k];
            ss += d * d;
        }
        ss = waveSum(ss);
        if (j == lane) muj = -ss;
    }

    float w  = __expf(muj * TEMP_INV - te);
    float S  = waveSum(w);
    float a  = w / (S + 1e-8f);
    float t2 = waveSum(a * muj);

    // exact mu_xy
    float ss = 0.0f;
#pragma unroll
    for (int k = 0; k < 4; ++k) {
        float d = (float)zh[(size_t)nb * D_EMB + lane * 4 + k] - zq[k];
        ss += d * d;
    }
    float mu_xy = -waveSum(ss);

    if (lane == 0) {
        float lamS = mu_xy;
        float d    = term1[i] + t2;              // lambda_T - lambda_S
        float sig  = 1.0f / (1.0f + __expf(-lamS));
        float pos  = -__logf(sig + 1e-8f);
        float cd   = core_values[nb] - core_values[q];
        float ad   = fabsf(d);
        float algn = (ad < 1.0f) ? 0.5f * d * d : ad - 0.5f;
        rowpart[i] = (pos + 0.1f * cd * cd + 0.1f * algn) * (1.0f / (float)PP);
    }
}

// ---------- K4: negatives + final reduction ----------
__global__ __launch_bounds__(256) void finalize_kernel(
    const _Float16* __restrict__ zh,
    const int* __restrict__ neg_idxs,
    const int* __restrict__ query_idx,
    const float* __restrict__ rowpart,
    float* __restrict__ out) {
    const int tid  = threadIdx.x;
    const int lane = tid & 31;
    const int wave = tid >> 5;
    const int q    = query_idx[0];
    const int nr   = neg_idxs[tid];

    float ss = 0.0f;
#pragma unroll 8
    for (int k = 0; k < D_EMB; ++k) {
        float d = (float)zh[(size_t)nr * D_EMB + k] - (float)zh[(size_t)q * D_EMB + k];
        ss += d * d;
    }
    float mu  = -ss;
    float sig = 1.0f / (1.0f + __expf(-mu));
    float neg = -__logf(1.0f - sig + 1e-8f);

    float v = rowpart[tid] + neg * (1.0f / (float)KK);
    v = waveSum(v);
    __shared__ float red[8];
    if (lane == 0) red[wave] = v;
    __syncthreads();
    if (tid == 0) {
        float s = 0.0f;
#pragma unroll
        for (int w = 0; w < 8; ++w) s += red[w];
        out[0] = s;
    }
}

extern "C" void kernel_launch(void* const* d_in, const int* in_sizes, int n_in,
                              void* d_out, int out_size, void* d_ws, size_t ws_size,
                              hipStream_t stream) {
    const float* z            = (const float*)d_in[0];
    const float* edge_times   = (const float*)d_in[1];
    const float* current_time = (const float*)d_in[2];
    const float* core_values  = (const float*)d_in[3];
    const float* omega        = (const float*)d_in[4];
    const float* phi          = (const float*)d_in[5];
    const int*   query_idx    = (const int*)d_in[6];
    const int*   neg_idxs     = (const int*)d_in[7];
    const int*   neighbor_idxs= (const int*)d_in[8];
    const int*   neighbors    = (const int*)d_in[9];
    float*       out          = (float*)d_out;

    // workspace: [ zh f16 N*D | term1 f32 P | rowpart f32 P ]
    _Float16* zh      = (_Float16*)d_ws;
    float*    term1   = (float*)((char*)d_ws + (size_t)NN * D_EMB * sizeof(_Float16));
    float*    rowpart = term1 + PP;

    normalize_kernel<<<NN / 8, 256, 0, stream>>>(z, zh);
    term1_kernel<<<1, 256, 0, stream>>>(zh, edge_times, current_time, omega, phi,
                                        query_idx, neighbor_idxs, neighbors, term1);
    term2_kernel<<<PP, 32, 0, stream>>>(zh, edge_times, current_time, core_values,
                                        omega, phi, query_idx, neighbor_idxs,
                                        neighbors, term1, rowpart);
    finalize_kernel<<<1, 256, 0, stream>>>(zh, neg_idxs, query_idx, rowpart, out);
}
